// MambaBlock_22806276342346
// MI455X (gfx1250) — matrix operand order repaired and checked by reading, not compiled, not run on hardware
//
#include <hip/hip_runtime.h>
#include <stdint.h>

#define D_MODEL 1024
#define D_STATE 16
#define D_CONV  4
#define D_INNER 1024
#define BATCH   2
#define SEQ     2048
#define MROWS   (BATCH * SEQ)   // 4096 GEMM rows

typedef __attribute__((ext_vector_type(16))) __bf16 bf16x16;
typedef __attribute__((ext_vector_type(8)))  __bf16 bf16x8;
typedef __attribute__((ext_vector_type(4)))  __bf16 bf16x4;
typedef __attribute__((ext_vector_type(8)))  float  f32x8;

__device__ __forceinline__ __bf16 f2bf(float f) {
    union { float f; uint32_t u; } c; c.f = f;
    uint32_t r = (c.u + 0x7FFFu + ((c.u >> 16) & 1u)) >> 16;   // RNE
    union { unsigned short s; __bf16 b; } o; o.s = (unsigned short)r;
    return o.b;
}

__device__ __forceinline__ float siluf_(float x)     { return x / (1.f + __expf(-x)); }
__device__ __forceinline__ float softplusf_(float x) { return x > 20.f ? x : log1pf(__expf(x)); }

// ---------------------------------------------------------------------------
// One-shot fp32 -> bf16 conversion (vectorized float4 -> 4x bf16).
// ---------------------------------------------------------------------------
__global__ __launch_bounds__(256) void f32_to_bf16_kernel(
    const float* __restrict__ src, __bf16* __restrict__ dst, int n4)
{
    int i = blockIdx.x * blockDim.x + threadIdx.x;
    if (i >= n4) return;
    float4 v = ((const float4*)src)[i];
    bf16x4 o; o[0] = f2bf(v.x); o[1] = f2bf(v.y); o[2] = f2bf(v.z); o[3] = f2bf(v.w);
    ((bf16x4*)dst)[i] = o;
}

// ---------------------------------------------------------------------------
// Async global -> LDS tile copy (CDNA5 ASYNCcnt path).
// One 128x32 bf16 A tile + one 128x32 bf16 B tile, rows padded to LDT=40.
// 256 threads x 4 chunks of 16B each (per-lane global_load_async_to_lds_b128).
// ---------------------------------------------------------------------------
__device__ __forceinline__ void async_tile_copy(
    const __bf16* __restrict__ A, const __bf16* __restrict__ W,
    int m0, int n0, int K, int k0,
    uint32_t ldsA, uint32_t ldsB, int tid)
{
    constexpr int LDT = 40;
#pragma unroll
    for (int p = 0; p < 2; ++p) {
        int c   = tid + p * 256;           // 0..511
        int row = c >> 2;                  // 0..127
        int kc  = c & 3;                   // 0..3 (8 bf16 = 16B per chunk)
        uint64_t ga = (uint64_t)(uintptr_t)(A + (size_t)(m0 + row) * K + k0 + kc * 8);
        uint32_t la = ldsA + (uint32_t)(row * LDT + kc * 8) * 2u;
        asm volatile("global_load_async_to_lds_b128 %0, %1, off"
                     :: "v"(la), "v"(ga) : "memory");
        uint64_t gw = (uint64_t)(uintptr_t)(W + (size_t)(n0 + row) * K + k0 + kc * 8);
        uint32_t lb = ldsB + (uint32_t)(row * LDT + kc * 8) * 2u;
        asm volatile("global_load_async_to_lds_b128 %0, %1, off"
                     :: "v"(lb), "v"(gw) : "memory");
    }
}

// ---------------------------------------------------------------------------
// bf16 WMMA GEMM:  C[M,N] = epi( A[M,K] @ W[N,K]^T + bias ), fp32 out.
// Block tile 128x128, BK=32, 8 waves, wave tile 64x32 (8 accumulators).
// Double-buffered LDS fed by global_load_async_to_lds_b128.
// EPI: 0 = none, 1 = softplus.
// ---------------------------------------------------------------------------
template <int EPI>
__global__ __launch_bounds__(256) void wmma_gemm_bf16(
    const __bf16* __restrict__ A, const __bf16* __restrict__ W,
    const float* __restrict__ bias, float* __restrict__ C,
    int M, int N, int K)
{
    constexpr int BM = 128, BN = 128, BK = 32, LDT = BK + 8;   // 40 bf16 = 80B rows
    constexpr int BUF = (BM + BN) * LDT;                       // elements per buffer
    __shared__ __align__(16) __bf16 smem[2 * BUF];

    const int tid   = threadIdx.x;
    const int lane  = tid & 31;
    const int wave  = tid >> 5;
    const int waveM = wave & 1;
    const int waveN = wave >> 1;
    const int m0 = blockIdx.y * BM;
    const int n0 = blockIdx.x * BN;
    const int half = lane >> 4;
    const int l16  = lane & 15;

    // LDS byte base (instruction-level ptrtoint; no static-init aggregates)
    const uint32_t lds0 = (uint32_t)(uintptr_t)smem;

    f32x8 acc[4][2];
#pragma unroll
    for (int mi = 0; mi < 4; ++mi)
#pragma unroll
        for (int ni = 0; ni < 2; ++ni)
#pragma unroll
            for (int e = 0; e < 8; ++e) acc[mi][ni][e] = 0.f;

    // prologue: async-stage tile 0 into buffer 0
    async_tile_copy(A, W, m0, n0, K, 0,
                    lds0, lds0 + (uint32_t)(BM * LDT * 2), tid);
    asm volatile("s_wait_asynccnt 0x0" ::: "memory");
    __syncthreads();

    const int nk = K / BK;
    for (int ki = 0; ki < nk; ++ki) {
        const int buf = ki & 1;
        const uint32_t nbase = lds0 + (uint32_t)(buf ^ 1) * (uint32_t)(BUF * 2);
        if (ki + 1 < nk)   // overlap: async-stage next tile into other buffer
            async_tile_copy(A, W, m0, n0, K, (ki + 1) * BK,
                            nbase, nbase + (uint32_t)(BM * LDT * 2), tid);

        const __bf16* sA = smem + buf * BUF;       // runtime-computed, no const array
        const __bf16* sB = sA + BM * LDT;

        bf16x16 afrag[4], bfrag[2];
#pragma unroll
        for (int mi = 0; mi < 4; ++mi) {
            int row = waveM * 64 + mi * 16 + l16;
            bf16x8 lo = *(const bf16x8*)&sA[row * LDT + half * 8];
            bf16x8 hi = *(const bf16x8*)&sA[row * LDT + 16 + half * 8];
#pragma unroll
            for (int e = 0; e < 8; ++e) { afrag[mi][e] = lo[e]; afrag[mi][8 + e] = hi[e]; }
        }
#pragma unroll
        for (int ni = 0; ni < 2; ++ni) {
            int col = waveN * 32 + ni * 16 + l16;
            bf16x8 lo = *(const bf16x8*)&sB[col * LDT + half * 8];
            bf16x8 hi = *(const bf16x8*)&sB[col * LDT + 16 + half * 8];
#pragma unroll
            for (int e = 0; e < 8; ++e) { bfrag[ni][e] = lo[e]; bfrag[ni][8 + e] = hi[e]; }
        }

#pragma unroll
        for (int mi = 0; mi < 4; ++mi)
#pragma unroll
            for (int ni = 0; ni < 2; ++ni)
                acc[mi][ni] = __builtin_amdgcn_wmma_f32_16x16x32_bf16(
                    false, afrag[mi], false, bfrag[ni],
                    (short)0, acc[mi][ni], false, false);

        asm volatile("s_wait_asynccnt 0x0" ::: "memory");
        __syncthreads();
    }

    // epilogue: C VGPR r holds M = r + 8*(lane>=16), N = lane%16
#pragma unroll
    for (int mi = 0; mi < 4; ++mi) {
#pragma unroll
        for (int ni = 0; ni < 2; ++ni) {
            int mbase = m0 + waveM * 64 + mi * 16 + half * 8;
            int ncol  = n0 + waveN * 32 + ni * 16 + l16;
            float bv = bias ? bias[ncol] : 0.f;
#pragma unroll
            for (int r = 0; r < 8; ++r) {
                float v = acc[mi][ni][r] + bv;
                if (EPI == 1) v = softplusf_(v);
                C[(size_t)(mbase + r) * N + ncol] = v;
            }
        }
    }
}

// ---------------------------------------------------------------------------
// Causal depthwise conv1d (K=4) + SiLU. Emits fp32 xc (for scan) and a bf16
// copy xcb (A-operand of the dt_proj GEMM).
// ---------------------------------------------------------------------------
__global__ __launch_bounds__(256) void conv_silu_kernel(
    const float* __restrict__ xr, const float* __restrict__ conv_w,
    const float* __restrict__ conv_b, float* __restrict__ xc,
    __bf16* __restrict__ xcb)
{
    int idx = blockIdx.x * blockDim.x + threadIdx.x;    // over MROWS*D_INNER
    if (idx >= MROWS * D_INNER) return;
    int d   = idx & (D_INNER - 1);
    int row = idx >> 10;                                // b*SEQ + l
    int l   = row & (SEQ - 1);

    float acc = conv_b[d];
#pragma unroll
    for (int k = 0; k < D_CONV; ++k) {
        int ls = l + k - (D_CONV - 1);
        if (ls >= 0)
            acc += conv_w[d * D_CONV + k] * xr[(size_t)(row + k - (D_CONV - 1)) * (2 * D_INNER) + d];
    }
    float v = siluf_(acc);
    xc[(size_t)row * D_INNER + d]  = v;
    xcb[(size_t)row * D_INNER + d] = f2bf(v);
}

// ---------------------------------------------------------------------------
// Selective SSM scan; one thread per (b,d), 16 states in VGPRs, serial in L.
// Fuses +xc*D skip and silu(res) gate; writes bf16 g (A of out_proj GEMM).
// ---------------------------------------------------------------------------
__global__ __launch_bounds__(256) void ssm_scan_kernel(
    const float* __restrict__ delta, const float* __restrict__ xc,
    const float* __restrict__ xr,    const float* __restrict__ A_log,
    const float* __restrict__ Bp,    const float* __restrict__ Cp,
    const float* __restrict__ Dp,    __bf16* __restrict__ gb)
{
    int t = blockIdx.x * blockDim.x + threadIdx.x;
    if (t >= BATCH * D_INNER) return;
    int b = t / D_INNER;
    int d = t % D_INNER;

    float An[D_STATE], Bn[D_STATE], Cn[D_STATE], h[D_STATE];
#pragma unroll
    for (int n = 0; n < D_STATE; ++n) {
        An[n] = -__expf(A_log[d * D_STATE + n]);
        Bn[n] = Bp[d * D_STATE + n];
        Cn[n] = Cp[d * D_STATE + n];
        h[n]  = 0.f;
    }
    float Dd = Dp[d];

    const float* dptr = delta + (size_t)b * SEQ * D_INNER + d;
    const float* xptr = xc    + (size_t)b * SEQ * D_INNER + d;
    const float* rptr = xr    + (size_t)b * SEQ * (2 * D_INNER) + D_INNER + d;
    __bf16*      gptr = gb    + (size_t)b * SEQ * D_INNER + d;

    for (int l = 0; l < SEQ; ++l) {
        float dt = dptr[(size_t)l * D_INNER];
        float xv = xptr[(size_t)l * D_INNER];
        float y = 0.f;
#pragma unroll
        for (int n = 0; n < D_STATE; ++n) {
            float dA = __expf(dt * An[n]);
            h[n] = dA * h[n] + dt * Bn[n] * xv;
            y += h[n] * Cn[n];
        }
        float r = rptr[(size_t)l * (2 * D_INNER)];
        gptr[(size_t)l * D_INNER] = f2bf((y + xv * Dd) * siluf_(r));
    }
}

// ---------------------------------------------------------------------------
extern "C" void kernel_launch(void* const* d_in, const int* in_sizes, int n_in,
                              void* d_out, int out_size, void* d_ws, size_t ws_size,
                              hipStream_t stream) {
    (void)in_sizes; (void)n_in; (void)out_size; (void)ws_size;
    const float* x      = (const float*)d_in[0];
    const float* W_in   = (const float*)d_in[1];
    const float* conv_w = (const float*)d_in[2];
    const float* conv_b = (const float*)d_in[3];
    const float* W_dt   = (const float*)d_in[4];
    const float* b_dt   = (const float*)d_in[5];
    const float* A_log  = (const float*)d_in[6];
    const float* Bp     = (const float*)d_in[7];
    const float* Cp     = (const float*)d_in[8];
    const float* Dp     = (const float*)d_in[9];
    const float* W_out  = (const float*)d_in[10];
    float* out = (float*)d_out;

    // ---- workspace layout (fp32 regions first, then bf16 regions) --------
    float*  xr    = (float*)d_ws;                                // 4096 x 2048 f32
    float*  xc    = xr    + (size_t)MROWS * 2 * D_INNER;         // 4096 x 1024 f32
    float*  delta = xc    + (size_t)MROWS * D_INNER;             // 4096 x 1024 f32
    __bf16* xb    = (__bf16*)(delta + (size_t)MROWS * D_INNER);  // 4096 x 1024 bf16
    __bf16* xcb   = xb    + (size_t)MROWS * D_INNER;             // 4096 x 1024 bf16
    __bf16* gbuf  = xcb   + (size_t)MROWS * D_INNER;             // 4096 x 1024 bf16
    __bf16* wbin  = gbuf  + (size_t)MROWS * D_INNER;             // 2048 x 1024 bf16
    __bf16* wbdt  = wbin  + (size_t)2 * D_INNER * D_MODEL;       // 1024 x 1024 bf16
    __bf16* wbout = wbdt  + (size_t)D_INNER * D_INNER;           // 1024 x 1024 bf16

    dim3 blk(256);

    // 0) one-shot fp32 -> bf16 conversions
    f32_to_bf16_kernel<<<(MROWS * D_MODEL / 4 + 255) / 256, blk, 0, stream>>>(x, xb, MROWS * D_MODEL / 4);
    f32_to_bf16_kernel<<<(2 * D_INNER * D_MODEL / 4 + 255) / 256, blk, 0, stream>>>(W_in, wbin, 2 * D_INNER * D_MODEL / 4);
    f32_to_bf16_kernel<<<(D_INNER * D_INNER / 4 + 255) / 256, blk, 0, stream>>>(W_dt, wbdt, D_INNER * D_INNER / 4);
    f32_to_bf16_kernel<<<(D_MODEL * D_INNER / 4 + 255) / 256, blk, 0, stream>>>(W_out, wbout, D_MODEL * D_INNER / 4);

    // 1) in_proj: xr = x @ W_in^T             (M=4096, N=2048, K=1024)
    wmma_gemm_bf16<0><<<dim3((2 * D_INNER) / 128, MROWS / 128), blk, 0, stream>>>(
        xb, wbin, nullptr, xr, MROWS, 2 * D_INNER, D_MODEL);

    // 2) causal depthwise conv + SiLU -> xc (f32) + xcb (bf16)
    conv_silu_kernel<<<(MROWS * D_INNER + 255) / 256, blk, 0, stream>>>(
        xr, conv_w, conv_b, xc, xcb);

    // 3) delta = softplus(xc @ W_dt^T + b_dt) (M=4096, N=1024, K=1024)
    wmma_gemm_bf16<1><<<dim3(D_INNER / 128, MROWS / 128), blk, 0, stream>>>(
        xcb, wbdt, b_dt, delta, MROWS, D_INNER, D_INNER);

    // 4) selective scan + skip + gate -> gbuf (bf16)
    ssm_scan_kernel<<<(BATCH * D_INNER + 255) / 256, blk, 0, stream>>>(
        delta, xc, xr, A_log, Bp, Cp, Dp, gbuf);

    // 5) out_proj: out = g @ W_out^T           (M=4096, N=1024, K=1024)
    wmma_gemm_bf16<0><<<dim3(D_MODEL / 128, MROWS / 128), blk, 0, stream>>>(
        gbuf, wbout, nullptr, out, MROWS, D_MODEL, D_MODEL);
}
